// GATLayer_10050223473241
// MI455X (gfx1250) — matrix operand order repaired and checked by reading, not compiled, not run on hardware
//
#include <hip/hip_runtime.h>

#define ALPHA 0.2f
#define BETA  0.3f
#define NEGV  -9.0e15f

typedef __attribute__((ext_vector_type(16))) __bf16 v16bf;
typedef __attribute__((ext_vector_type(8)))  float  v8f;
typedef unsigned short u16b;
typedef unsigned int   u32b;
typedef unsigned int u32x4 __attribute__((ext_vector_type(4)));
typedef int          i32x4 __attribute__((ext_vector_type(4)));
typedef int          i32x8 __attribute__((ext_vector_type(8)));

union V16U { v16bf v; __bf16 e[16]; uint4 q[2]; u16b b[16]; };

#if defined(__has_builtin)
#if __has_builtin(__builtin_amdgcn_tensor_load_to_lds) && \
    __has_builtin(__builtin_amdgcn_s_wait_tensorcnt)
#define USE_TDM 1
#endif
#endif

__device__ __forceinline__ u16b f2bf_bits(float f) {
  union { float f; u32b u; } x; x.f = f;
  u32b r = x.u + 0x7FFFu + ((x.u >> 16) & 1u);   // round-to-nearest-even
  return (u16b)(r >> 16);
}

__device__ __forceinline__ float redmax16(float v) {
  v = fmaxf(v, __shfl_xor(v, 1));
  v = fmaxf(v, __shfl_xor(v, 2));
  v = fmaxf(v, __shfl_xor(v, 4));
  v = fmaxf(v, __shfl_xor(v, 8));
  return v;
}
__device__ __forceinline__ float redsum16(float v) {
  v += __shfl_xor(v, 1);
  v += __shfl_xor(v, 2);
  v += __shfl_xor(v, 4);
  v += __shfl_xor(v, 8);
  return v;
}

#ifdef USE_TDM
// One 2-D TDM tile load: tile_d1 rows x tile_d0 elements (2B), row stride
// 'stride0' elements; LDS rows padded +16B (pad_amount=3) every 'pad_code'
// interval (code 4 = 32 DWORDs = 128B row, code 5 = 64 DWORDs = 256B row).
// D# fields per CDNA5 ISA (type=2, count=1).
__device__ __forceinline__ void tdm_load_2d(unsigned lds_off,
                                            const void* gsrc,
                                            unsigned tensor_d0, unsigned tensor_d1,
                                            unsigned tile_d0, unsigned tile_d1,
                                            unsigned stride0, unsigned pad_code) {
  unsigned long long ga = (unsigned long long)(size_t)gsrc;
  u32x4 g0;
  g0.x = 1u;                                     // count=1, user mode
  g0.y = lds_off;                                // D#.lds_addr
  g0.z = (unsigned)(ga & 0xffffffffu);           // global_addr[31:0]
  g0.w = (unsigned)((ga >> 32) & 0x1ffffffu) | (2u << 30);  // addr[56:32]|type=2
  i32x8 g1;
  g1[0] = (int)((1u << 16) |                     // data_size = 2B
                (1u << 20) |                     // pad_enable
                (pad_code << 22) |               // pad_interval
                (3u << 25));                     // pad_amount = 4 DWORDs (16B)
  g1[1] = (int)((tensor_d0 & 0xffffu) << 16);    // atomic_addr=0 | dim0[15:0]
  g1[2] = (int)(((tensor_d0 >> 16) & 0xffffu) | ((tensor_d1 & 0xffffu) << 16));
  g1[3] = (int)(((tensor_d1 >> 16) & 0xffffu) | ((tile_d0 & 0xffffu) << 16));
  g1[4] = (int)(tile_d1 & 0xffffu);              // tile_dim1 | tile_dim2=0
  g1[5] = (int)stride0;                          // tensor_dim0_stride[31:0]
  g1[6] = 0;
  g1[7] = 0;
  i32x4 gz4 = {};
#if __clang_major__ >= 23
  i32x8 gz8 = {};
  __builtin_amdgcn_tensor_load_to_lds(g0, g1, gz4, gz4, gz8, 0);
#else
  __builtin_amdgcn_tensor_load_to_lds(g0, g1, gz4, gz4, 0);
#endif
}
#endif

// ---------------------------------------------------------------------------
// Kernel 1: Wh = h @ w (bf16 WMMA, f32 accum).  Emits Wh (f32), Whb (bf16
// row-major), WhbT (bf16 col-major), s1[i] = sum_k Wh[i,k]*a1[i,k].
// grid: 64 blocks x 256 threads (8 waves x 16 rows = 128 rows/block)
// ---------------------------------------------------------------------------
__global__ void __launch_bounds__(256) gat_wh(const float* __restrict__ h,
                                              const float* __restrict__ w,
                                              const float* __restrict__ a,
                                              float* __restrict__ Wh,
                                              u16b*  __restrict__ Whb,
                                              u16b*  __restrict__ WhbT,
                                              float* __restrict__ s1) {
  // w staged TRANSPOSED (wT[col][k], pitch 136) so B-fragments are contiguous
  __shared__ __attribute__((aligned(16))) u16b wlds[128 * 136];

  const int tid = threadIdx.x;
  for (int t = tid; t < 128 * 16; t += 256) {
    int row = t >> 4, seg = t & 15;                 // w row k, 8-col segment
    const float* src = w + row * 128 + seg * 8;
#pragma unroll
    for (int i = 0; i < 8; ++i)
      wlds[(seg * 8 + i) * 136 + row] = f2bf_bits(src[i]);   // wT[col][k]
  }
  __syncthreads();

  const int lane = tid & 31;
  const int wave = tid >> 5;
  const int l15  = lane & 15;
  const int half = lane >> 4;
  const int i0   = blockIdx.x * 128 + wave * 16;

  // A-fragments of h rows (16x32 bf16, 4 K-blocks)
  const float* hrow = h + (size_t)(i0 + l15) * 128;
  V16U ha[4];
#pragma unroll
  for (int kb = 0; kb < 4; ++kb) {
#pragma unroll
    for (int e = 0; e < 16; ++e) {
      int K = kb * 32 + (e < 8 ? half * 8 + e : 16 + half * 8 + (e - 8));
      ha[kb].b[e] = f2bf_bits(hrow[K]);
    }
  }

  v8f c[8];
#pragma unroll
  for (int n = 0; n < 8; ++n) {
    v8f acc = {};
#pragma unroll
    for (int kb = 0; kb < 4; ++kb) {
      // B[k][n] = w[k][col]; from wT: contiguous along k
      const u16b* bp = &wlds[(n * 16 + l15) * 136 + kb * 32 + half * 8];
      V16U bfr;
      bfr.q[0] = *(const uint4*)(bp);
      bfr.q[1] = *(const uint4*)(bp + 16);
      acc = __builtin_amdgcn_wmma_f32_16x16x32_bf16(
          false, ha[kb].v, false, bfr.v, (short)0, acc, false, false);
    }
    c[n] = acc;
  }

  float sr[8];
#pragma unroll
  for (int r = 0; r < 8; ++r) sr[r] = 0.0f;

#pragma unroll
  for (int n = 0; n < 8; ++n) {
#pragma unroll
    for (int r = 0; r < 8; ++r) {
      int row = i0 + half * 8 + r;
      int col = n * 16 + l15;
      float v = c[n][r];
      u16b bb = f2bf_bits(v);
      Wh  [(size_t)row * 128 + col]  = v;
      Whb [(size_t)row * 128 + col]  = bb;
      WhbT[(size_t)col * 8192 + row] = bb;
      sr[r] += v * a[(size_t)row * 256 + col];    // a1 = a[:, :F, 0]
    }
  }
#pragma unroll
  for (int r = 0; r < 8; ++r) {
    float v = redsum16(sr[r]);
    if (l15 == 0) s1[i0 + half * 8 + r] = v;
  }
}

// ---------------------------------------------------------------------------
// Kernel 2: fused masked attention (flash-style online softmax) + epilogue.
// grid: 128 blocks x 128 threads (4 waves x 16 rows = 64 rows/block).
// 128 panels of 64 j-columns; DOUBLE-BUFFERED dual LDS panels (row-major for
// the score matmul, col-major for the accumulate matmul).  TDM loads panel
// jp+1 while all waves compute on panel jp; one barrier per panel both
// publishes panel jp and retires panel jp-1's buffer.
// ---------------------------------------------------------------------------
__global__ void __launch_bounds__(128) gat_attn(const float* __restrict__ a,
                                                const int*   __restrict__ adj,
                                                const float* __restrict__ Wh,
                                                const u16b*  __restrict__ Whb,
                                                const u16b*  __restrict__ WhbT,
                                                const float* __restrict__ s1,
                                                float* __restrict__ out) {
  __shared__ __attribute__((aligned(16))) u16b panelRM[2][64 * 136]; // [j][k]
  __shared__ __attribute__((aligned(16))) u16b panelCM[2][128 * 72]; // [k][j]
  __shared__ __attribute__((aligned(16))) u16b pbuf[4][16 * 32];     // P scratch

  const int tid  = threadIdx.x;
  const int lane = tid & 31;
  const int wave = tid >> 5;
  const int l15  = lane & 15;
  const int half = lane >> 4;
  const int i0   = blockIdx.x * 64 + wave * 16;

  // a2 A-fragments (a2 = a[:, F:, 0])
  const float* a2row = a + (size_t)(i0 + l15) * 256 + 128;
  V16U a2f[4];
#pragma unroll
  for (int kb = 0; kb < 4; ++kb) {
#pragma unroll
    for (int e = 0; e < 16; ++e) {
      int K = kb * 32 + (e < 8 ? half * 8 + e : 16 + half * 8 + (e - 8));
      a2f[kb].b[e] = f2bf_bits(a2row[K]);
    }
  }

  float m[8], lsum[8], s1r[8];
  v8f acc[8];
#pragma unroll
  for (int r = 0; r < 8; ++r) {
    m[r] = -__builtin_inff();
    lsum[r] = 0.0f;
    s1r[r] = s1[i0 + half * 8 + r];
  }
#pragma unroll
  for (int n = 0; n < 8; ++n) acc[n] = (v8f){};

#ifdef USE_TDM
  // prologue: start panel 0 into buffer 0
  if (__builtin_amdgcn_readfirstlane(wave) == 0) {
    tdm_load_2d((unsigned)(size_t)&panelRM[0][0], Whb,
                128u, 8192u, 128u, 64u, 128u, 5u);
    tdm_load_2d((unsigned)(size_t)&panelCM[0][0], WhbT,
                8192u, 128u, 64u, 128u, 8192u, 4u);
  }
#endif

#pragma unroll 1
  for (int jp = 0; jp < 128; ++jp) {
    const int cur = jp & 1;
#ifdef USE_TDM
    if (__builtin_amdgcn_readfirstlane(wave) == 0)
      __builtin_amdgcn_s_wait_tensorcnt(0);      // panel jp landed in LDS
    __syncthreads();                              // publish jp; retire jp-1 buf
    if (jp + 1 < 128 && __builtin_amdgcn_readfirstlane(wave) == 0) {
      tdm_load_2d((unsigned)(size_t)&panelRM[cur ^ 1][0],
                  Whb + (size_t)(jp + 1) * 64 * 128,
                  128u, 8192u, 128u, 64u, 128u, 5u);
      tdm_load_2d((unsigned)(size_t)&panelCM[cur ^ 1][0],
                  WhbT + (size_t)(jp + 1) * 64,
                  8192u, 128u, 64u, 128u, 8192u, 4u);
    }
#else
    __syncthreads();                              // retire old contents of cur
    for (int t = tid; t < 64 * 16; t += 128) {    // RM: 64 rows x 16 chunks
      int row = t >> 4, seg = t & 15;
      *(uint4*)&panelRM[cur][row * 136 + seg * 8] =
          *(const uint4*)&Whb[(size_t)(jp * 64 + row) * 128 + seg * 8];
    }
    for (int t = tid; t < 128 * 8; t += 128) {    // CM: 128 cols x 8 chunks
      int col = t >> 3, seg = t & 7;
      *(uint4*)&panelCM[cur][col * 72 + seg * 8] =
          *(const uint4*)&WhbT[(size_t)col * 8192 + jp * 64 + seg * 8];
    }
    __syncthreads();
#endif
    const u16b* pRM = &panelRM[cur][0];
    const u16b* pCM = &panelCM[cur][0];

#pragma unroll
    for (int jpair = 0; jpair < 2; ++jpair) {
#pragma unroll
      for (int sub = 0; sub < 2; ++sub) {
        const int jj    = jpair * 2 + sub;
        const int jglob = jp * 64 + jj * 16;

        __builtin_prefetch(&adj[(size_t)(i0 + half * 8) * 8192 + jglob + 256], 0, 0);

        // e-tile = a2(16x128) @ Whb_j(16x128)^T ; B[k][n] = Whb[j0+n][k]
        v8f ef = {};
        const u16b* brow = pRM + (jj * 16 + l15) * 136;
#pragma unroll
        for (int kb = 0; kb < 4; ++kb) {
          V16U bfr;
          bfr.q[0] = *(const uint4*)(brow + kb * 32 + half * 8);
          bfr.q[1] = *(const uint4*)(brow + kb * 32 + half * 8 + 16);
          ef = __builtin_amdgcn_wmma_f32_16x16x32_bf16(
              false, a2f[kb].v, false, bfr.v, (short)0, ef, false, false);
        }

        // bias + LeakyReLU + adjacency mask (non-temporal: one-shot stream)
        float p[8];
#pragma unroll
        for (int r = 0; r < 8; ++r) {
          float x = ef[r] + s1r[r];
          x = x > 0.0f ? x : ALPHA * x;
          int ad = __builtin_nontemporal_load(
              &adj[(size_t)(i0 + half * 8 + r) * 8192 + jglob + l15]);
          p[r] = ad > 0 ? x : NEGV;
        }

        // online softmax update per row
#pragma unroll
        for (int r = 0; r < 8; ++r) {
          float newm  = fmaxf(m[r], redmax16(p[r]));
          float scale = __expf(m[r] - newm);
          float pv    = __expf(p[r] - newm);
          lsum[r] = lsum[r] * scale + redsum16(pv);
          m[r]    = newm;
          p[r]    = pv;
#pragma unroll
          for (int n = 0; n < 8; ++n) acc[n][r] *= scale;
        }

        // stash P (bf16) to per-wave LDS scratch, column half = sub
        u16b* pb = &pbuf[wave][0];
#pragma unroll
        for (int r = 0; r < 8; ++r)
          pb[(half * 8 + r) * 32 + sub * 16 + l15] = f2bf_bits(p[r]);
      }

      // same-wave LDS RAW: drain DS ops, block compiler reordering
      asm volatile("s_wait_dscnt 0" ::: "memory");

      // reload P as a 16x32 A-fragment
      V16U pa;
      const u16b* prow = &pbuf[wave][l15 * 32];
      pa.q[0] = *(const uint4*)(prow + half * 8);
      pa.q[1] = *(const uint4*)(prow + 16 + half * 8);

      // acc(16x128) += P(16x32) @ Whb_pair(32x128); B from col-major panel
      const int jb = jpair * 32;
#pragma unroll
      for (int n = 0; n < 8; ++n) {
        const u16b* bp = pCM + (n * 16 + l15) * 72 + jb + half * 8;
        V16U bfr;
        bfr.q[0] = *(const uint4*)(bp);
        bfr.q[1] = *(const uint4*)(bp + 16);
        acc[n] = __builtin_amdgcn_wmma_f32_16x16x32_bf16(
            false, pa.v, false, bfr.v, (short)0, acc[n], false, false);
      }
    }
  }

  // epilogue: h' = acc / lsum; out = elu(h' + BETA*Wh)
#pragma unroll
  for (int r = 0; r < 8; ++r) lsum[r] = 1.0f / lsum[r];
#pragma unroll
  for (int n = 0; n < 8; ++n) {
#pragma unroll
    for (int r = 0; r < 8; ++r) {
      int row = i0 + half * 8 + r;
      int col = n * 16 + l15;
      float x = acc[n][r] * lsum[r] + BETA * Wh[(size_t)row * 128 + col];
      out[(size_t)row * 128 + col] = x > 0.0f ? x : (__expf(x) - 1.0f);
    }
  }
}

// ---------------------------------------------------------------------------
extern "C" void kernel_launch(void* const* d_in, const int* in_sizes, int n_in,
                              void* d_out, int out_size, void* d_ws, size_t ws_size,
                              hipStream_t stream) {
  (void)in_sizes; (void)n_in; (void)out_size; (void)ws_size;
  const float* h   = (const float*)d_in[0];
  const int*   adj = (const int*)d_in[1];
  // d_in[2] = batch_i (unused by the reference math)
  const float* w   = (const float*)d_in[3];
  const float* a   = (const float*)d_in[4];
  float* out = (float*)d_out;

  char* ws = (char*)d_ws;
  float* Wh   = (float*)ws;                              // 4 MB
  u16b*  Whb  = (u16b*)(ws + (size_t)8192 * 128 * 4);    // 2 MB
  u16b*  WhbT = (u16b*)(ws + (size_t)8192 * 128 * 6);    // 2 MB
  float* s1   = (float*)(ws + (size_t)8192 * 128 * 8);   // 32 KB

  gat_wh  <<<64, 256, 0, stream>>>(h, w, a, Wh, Whb, WhbT, s1);
  gat_attn<<<128, 128, 0, stream>>>(a, adj, Wh, Whb, WhbT, s1, out);
}
